// GroupedQueryAttention_46763603919522
// MI455X (gfx1250) — compile-verified, matrix operand order checked by reading, not run on hardware
//
#include <hip/hip_runtime.h>
#include <hip/hip_bf16.h>
#include <math.h>
#include <stdint.h>

#define DDIM 1024
#define SEQ  4096
#define GRP  16
#define HDIM 64

typedef __attribute__((ext_vector_type(16))) __bf16 bf16x16;
typedef __attribute__((ext_vector_type(8)))  __bf16 bf16x8;
typedef __attribute__((ext_vector_type(4)))  __bf16 bf16x4;
typedef __attribute__((ext_vector_type(8)))  float  f32x8;
typedef __attribute__((ext_vector_type(4)))  float  f32x4;
typedef int v4i __attribute__((vector_size(4 * sizeof(int))));

static __device__ __forceinline__ f32x8 wmma_bf16(bf16x16 a, bf16x16 b, f32x8 c) {
  // D = A(16x32 bf16) * B(32x16 bf16) + C(16x16 f32)
  return __builtin_amdgcn_wmma_f32_16x16x32_bf16(false, a, false, b, (short)0, c,
                                                 false, false);
}

// Load one 16x32 bf16 A/B fragment: lanes 0-15 hold K=0..7 / 16..23,
// lanes 16-31 hold K=8..15 / 24..31 (caller adds (lane>>4)*8).
static __device__ __forceinline__ bf16x16 frag_ld(const __bf16* p) {
  bf16x8 lo = *(const bf16x8*)(p);
  bf16x8 hi = *(const bf16x8*)(p + 16);
  bf16x16 r;
#pragma unroll
  for (int i = 0; i < 8; ++i) { r[i] = lo[i]; r[i + 8] = hi[i]; }
  return r;
}

// ---- CDNA5 async global->LDS copy (ASYNCcnt), 16 bytes per lane ------------
static __device__ __forceinline__ void async_cp16(const void* g, void* l) {
#if __has_builtin(__builtin_amdgcn_global_load_async_to_lds_b128)
  __builtin_amdgcn_global_load_async_to_lds_b128(
      (__attribute__((address_space(1))) v4i*)(uintptr_t)g,
      (__attribute__((address_space(3))) v4i*)(uintptr_t)l, 0, 0);
#else
  asm volatile("global_load_async_to_lds_b128 %0, %1, off"
               :: "v"((unsigned)(uintptr_t)l),
                  "v"((unsigned long long)(uintptr_t)g)
               : "memory");
#endif
}

static __device__ __forceinline__ void wait_async0() {
#if __has_builtin(__builtin_amdgcn_s_wait_asynccnt)
  __builtin_amdgcn_s_wait_asynccnt(0);
#else
  asm volatile("s_wait_asynccnt 0x0" ::: "memory");
#endif
}

// ---- 16-lane XOR butterfly via v_permlane16_b32 (pure VALU) ----------------
static constexpr unsigned xor_sel(int m, int base) {
  unsigned s = 0;
  for (int j = 0; j < 8; ++j) s |= (unsigned)(((base + j) ^ m) & 15) << (4 * j);
  return s;
}

template <int M>
static __device__ __forceinline__ float shfl_xor16(float x) {
#if __has_builtin(__builtin_amdgcn_permlane16)
  constexpr unsigned s0 = xor_sel(M, 0);
  constexpr unsigned s1 = xor_sel(M, 8);
  unsigned u = __builtin_bit_cast(unsigned, x);
  u = __builtin_amdgcn_permlane16(u, u, s0, s1, false, false);
  return __builtin_bit_cast(float, u);
#else
  return __shfl_xor(x, M, 32);
#endif
}

// ---------------------------------------------------------------------------
// fp32 -> bf16 conversion (weights and activations)
// ---------------------------------------------------------------------------
__global__ __launch_bounds__(256) void cvt_f32_to_bf16(const float* __restrict__ src,
                                                       __bf16* __restrict__ dst,
                                                       int n) {
  int i = (blockIdx.x * 256 + threadIdx.x) * 4;
  if (i + 3 < n) {
    f32x4 v = *(const f32x4*)(src + i);
    bf16x4 o;
#pragma unroll
    for (int j = 0; j < 4; ++j) o[j] = (__bf16)v[j];
    *(bf16x4*)(dst + i) = o;
  }
}

// ---------------------------------------------------------------------------
// GEMM: C[4096,1024] = X[4096,1024] @ W.T + b (both bf16, NT, async staged)
// Block 128x128x32, 256 threads = 8 waves, wave tile 32x64 (2x4 WMMA),
// double-buffered LDS, one barrier per K step.
// ---------------------------------------------------------------------------
#define BM 128
#define BN 128
#define BK 32
#define LDT 40  // padded LDS row stride (elements): 80 B, 16B-aligned

enum { GM_QK = 0, GM_VT = 1, GM_OUT = 2 };

template <int MODE>
__global__ __launch_bounds__(256) void gemm_bf16_wmma(
    const __bf16* __restrict__ x, const __bf16* __restrict__ w,
    const float* __restrict__ bias, void* __restrict__ outp, float oscale) {
  __shared__ __bf16 lA[2][BM * LDT];
  __shared__ __bf16 lB[2][BN * LDT];

  const int tid  = threadIdx.x;
  const int lane = tid & 31, wave = tid >> 5;
  const int fr   = lane & 15;
  const int kb8  = (lane >> 4) << 3;
  const int hb   = kb8;
  const int wm   = (wave & 3) * 32;
  const int wn   = (wave >> 2) * 64;
  const int m0   = blockIdx.y * BM;
  const int n0   = blockIdx.x * BN;

  const int srow  = tid >> 1;        // staging row 0..127
  const int shalf = (tid & 1) * 16;  // staging K offset 0/16

  const __bf16* gx = x + (size_t)(m0 + srow) * DDIM + shalf;
  const __bf16* gw = w + (size_t)(n0 + srow) * DDIM + shalf;
  __bf16* sa = &lA[0][srow * LDT + shalf];
  __bf16* sb = &lB[0][srow * LDT + shalf];

  f32x8 acc[2][4];
#pragma unroll
  for (int mi = 0; mi < 2; ++mi)
#pragma unroll
    for (int ni = 0; ni < 4; ++ni)
#pragma unroll
      for (int r = 0; r < 8; ++r) acc[mi][ni][r] = 0.f;

  auto stage = [&](int kt, int buf) {
    const int bo = buf * (BM * LDT);
    async_cp16(gx + kt,     sa + bo);
    async_cp16(gx + kt + 8, sa + bo + 8);
    async_cp16(gw + kt,     sb + bo);
    async_cp16(gw + kt + 8, sb + bo + 8);
  };

  stage(0, 0);
  int cur = 0;
  for (int kt = 0; kt < DDIM; kt += BK) {
    wait_async0();
    __syncthreads();
    if (kt + BK < DDIM) stage(kt + BK, cur ^ 1);

    bf16x16 af[2], bw[4];
#pragma unroll
    for (int mi = 0; mi < 2; ++mi)
      af[mi] = frag_ld(&lA[cur][(wm + mi * 16 + fr) * LDT + kb8]);
#pragma unroll
    for (int ni = 0; ni < 4; ++ni)
      bw[ni] = frag_ld(&lB[cur][(wn + ni * 16 + fr) * LDT + kb8]);
#pragma unroll
    for (int mi = 0; mi < 2; ++mi)
#pragma unroll
      for (int ni = 0; ni < 4; ++ni)
        acc[mi][ni] = wmma_bf16(af[mi], bw[ni], acc[mi][ni]);
    cur ^= 1;
  }

  // epilogue: bias (+ optional scale folded into Q) + layout scatter
#pragma unroll
  for (int mi = 0; mi < 2; ++mi) {
#pragma unroll
    for (int ni = 0; ni < 4; ++ni) {
      const int   n  = n0 + wn + ni * 16 + fr;
      const float bn = bias[n];
#pragma unroll
      for (int r = 0; r < 8; ++r) {
        const int   m   = m0 + wm + mi * 16 + r + hb;
        const float val = acc[mi][ni][r] + bn;
        if (MODE == GM_OUT) {
          ((float*)outp)[(size_t)m * DDIM + n] = val;
        } else if (MODE == GM_QK) {  // per-head [g][s][dh]
          const int gq = n >> 6, dh = n & 63;
          ((__bf16*)outp)[((size_t)gq * SEQ + m) * HDIM + dh] = (__bf16)(val * oscale);
        } else {                     // GM_VT: transposed [g][dh][s]
          const int gq = n >> 6, dh = n & 63;
          ((__bf16*)outp)[((size_t)gq * HDIM + dh) * SEQ + m] = (__bf16)val;
        }
      }
    }
  }
}

// ---------------------------------------------------------------------------
// Flash attention: grid (S/128, G), 8 waves, 16 queries/wave, KV blocks of 32.
// Async double-buffered K/V staging; one barrier per KV step; softmax scale
// pre-folded into Q; permlane16 row reductions.
// ---------------------------------------------------------------------------
#define LKS 72  // K tile row stride: 144 B, 16B-aligned

__global__ __launch_bounds__(256) void attn_wmma(
    const __bf16* __restrict__ qb, const __bf16* __restrict__ kb,
    const __bf16* __restrict__ vtb, const int* __restrict__ mask,
    __bf16* __restrict__ outp) {
  __shared__ __bf16 lK[2][32 * LKS];     // [key][dh]
  __shared__ __bf16 lV[2][64 * LDT];     // [dh][key] (V pre-transposed)
  __shared__ __bf16 lP[8][16 * LDT];     // per-wave P tile [q][key]

  const int g    = blockIdx.y;
  const int tid  = threadIdx.x;
  const int lane = tid & 31, wave = tid >> 5;
  const int fr   = lane & 15;
  const int kb8  = (lane >> 4) << 3;
  const int hb   = kb8;
  const int q0   = blockIdx.x * 128 + wave * 16;

  // Q fragments (already scaled by 1/sqrt(DH)) held for the whole KV loop
  const __bf16* qrow = qb + ((size_t)g * SEQ + q0 + fr) * HDIM;
  const bf16x16 aq0 = frag_ld(qrow + kb8);
  const bf16x16 aq1 = frag_ld(qrow + 32 + kb8);

  f32x8 acco[4];
  float rm[8], rl[8];
#pragma unroll
  for (int ni = 0; ni < 4; ++ni)
#pragma unroll
    for (int r = 0; r < 8; ++r) acco[ni][r] = 0.f;
#pragma unroll
  for (int r = 0; r < 8; ++r) { rm[r] = -__builtin_inff(); rl[r] = 0.f; }

  const int skey = tid >> 3, sdh = (tid & 7) * 8;  // K staging map (32x64)
  const int vdh  = tid >> 2, vko = (tid & 3) * 8;  // V staging map (64x32)
  const __bf16* gk = kb + ((size_t)g * SEQ + skey) * HDIM + sdh;
  const __bf16* gv = vtb + ((size_t)g * HDIM + vdh) * SEQ + vko;
  __bf16* sk = &lK[0][skey * LKS + sdh];
  __bf16* sv = &lV[0][vdh * LDT + vko];

  auto stage = [&](int kv0, int buf) {
    async_cp16(gk + (size_t)kv0 * HDIM, sk + buf * (32 * LKS));
    async_cp16(gv + kv0,               sv + buf * (64 * LDT));
  };

  stage(0, 0);
  int cur = 0;
  for (int kv0 = 0; kv0 < SEQ; kv0 += 32) {
    wait_async0();
    __syncthreads();
    if (kv0 + 32 < SEQ) stage(kv0 + 32, cur ^ 1);

    // scores: S[16q x 32keys] = Q (16x64) . K^T
    f32x8 s0, s1;
#pragma unroll
    for (int r = 0; r < 8; ++r) { s0[r] = 0.f; s1[r] = 0.f; }
    {
      const __bf16* kbuf = &lK[cur][0];
      bf16x16 b00 = frag_ld(&kbuf[(fr)      * LKS + kb8]);
      bf16x16 b01 = frag_ld(&kbuf[(fr)      * LKS + 32 + kb8]);
      bf16x16 b10 = frag_ld(&kbuf[(16 + fr) * LKS + kb8]);
      bf16x16 b11 = frag_ld(&kbuf[(16 + fr) * LKS + 32 + kb8]);
      s0 = wmma_bf16(aq0, b00, s0);
      s0 = wmma_bf16(aq1, b01, s0);
      s1 = wmma_bf16(aq0, b10, s1);
      s1 = wmma_bf16(aq1, b11, s1);
    }

    // key mask (scale already folded into Q)
    const int mk0 = mask[kv0 + fr];
    const int mk1 = mask[kv0 + 16 + fr];
#pragma unroll
    for (int r = 0; r < 8; ++r) {
      s0[r] = mk0 ? s0[r] : -__builtin_inff();
      s1[r] = mk1 ? s1[r] : -__builtin_inff();
    }

    // online softmax; row r lives across the 16 lanes of this half-wave
#pragma unroll
    for (int r = 0; r < 8; ++r) {
      float t = fmaxf(s0[r], s1[r]);
      t = fmaxf(t, shfl_xor16<1>(t));
      t = fmaxf(t, shfl_xor16<2>(t));
      t = fmaxf(t, shfl_xor16<4>(t));
      t = fmaxf(t, shfl_xor16<8>(t));
      const float mnew = fmaxf(rm[r], t);
      const float corr = __expf(rm[r] - mnew);
      const float p0 = __expf(s0[r] - mnew);
      const float p1 = __expf(s1[r] - mnew);
      float rs = p0 + p1;
      rs += shfl_xor16<1>(rs);
      rs += shfl_xor16<2>(rs);
      rs += shfl_xor16<4>(rs);
      rs += shfl_xor16<8>(rs);
      rl[r] = rl[r] * corr + rs;
      rm[r] = mnew;
#pragma unroll
      for (int ni = 0; ni < 4; ++ni) acco[ni][r] *= corr;
      s0[r] = p0;
      s1[r] = p1;
    }

    // C-layout -> A-layout for P via wave-private LDS (in-order per wave)
    __bf16* pw = &lP[wave][0];
#pragma unroll
    for (int r = 0; r < 8; ++r) {
      pw[(r + hb) * LDT + fr]      = (__bf16)s0[r];
      pw[(r + hb) * LDT + 16 + fr] = (__bf16)s1[r];
    }
    asm volatile("" ::: "memory");  // keep ds stores before ds frag loads

    const bf16x16 ap = frag_ld(&pw[fr * LDT + kb8]);
#pragma unroll
    for (int ni = 0; ni < 4; ++ni) {
      bf16x16 bv = frag_ld(&lV[cur][(ni * 16 + fr) * LDT + kb8]);
      acco[ni] = wmma_bf16(ap, bv, acco[ni]);
    }
    cur ^= 1;
  }

  // normalize and write bf16 attention output [s][g*64+dh]
#pragma unroll
  for (int ni = 0; ni < 4; ++ni) {
#pragma unroll
    for (int r = 0; r < 8; ++r) {
      const int srow = q0 + r + hb;
      const int dcol = g * HDIM + ni * 16 + fr;
      outp[(size_t)srow * DDIM + dcol] = (__bf16)(acco[ni][r] / rl[r]);
    }
  }
}

// ---------------------------------------------------------------------------
extern "C" void kernel_launch(void* const* d_in, const int* in_sizes, int n_in,
                              void* d_out, int out_size, void* d_ws, size_t ws_size,
                              hipStream_t stream) {
  const float* q    = (const float*)d_in[0];
  const float* k    = (const float*)d_in[1];
  const float* v    = (const float*)d_in[2];
  const int*   mask = (const int*)d_in[3];
  const float* Wq   = (const float*)d_in[4];
  const float* bq   = (const float*)d_in[5];
  const float* Wk   = (const float*)d_in[6];
  const float* bk   = (const float*)d_in[7];
  const float* Wv   = (const float*)d_in[8];
  const float* bv   = (const float*)d_in[9];
  const float* Wo   = (const float*)d_in[10];
  const float* bo   = (const float*)d_in[11];

  // workspace layout (bf16 elements); total 32M elems = 64 MB
  __bf16* ws  = (__bf16*)d_ws;
  const size_t MEL = (size_t)1024 * 1024;
  __bf16* wqb = ws + 0 * MEL;
  __bf16* wkb = ws + 1 * MEL;
  __bf16* wvb = ws + 2 * MEL;
  __bf16* wob = ws + 3 * MEL;
  __bf16* xq  = ws + 4 * MEL;    // query bf16 [S][D]
  __bf16* xk  = ws + 8 * MEL;    // key   bf16 [S][D]
  __bf16* xv  = ws + 12 * MEL;   // value bf16 [S][D]
  __bf16* qbf = ws + 16 * MEL;   // [G][S][DH] (pre-scaled by 1/8)
  __bf16* kbf = ws + 20 * MEL;   // [G][S][DH]
  __bf16* vtb = ws + 24 * MEL;   // [G][DH][S]
  __bf16* abf = ws + 28 * MEL;   // [S][D]

  const int WN = DDIM * DDIM;        // 1M
  const int AN = SEQ * DDIM;         // 4M
  cvt_f32_to_bf16<<<WN / 1024, 256, 0, stream>>>(Wq, wqb, WN);
  cvt_f32_to_bf16<<<WN / 1024, 256, 0, stream>>>(Wk, wkb, WN);
  cvt_f32_to_bf16<<<WN / 1024, 256, 0, stream>>>(Wv, wvb, WN);
  cvt_f32_to_bf16<<<WN / 1024, 256, 0, stream>>>(Wo, wob, WN);
  cvt_f32_to_bf16<<<AN / 1024, 256, 0, stream>>>(q, xq, AN);
  cvt_f32_to_bf16<<<AN / 1024, 256, 0, stream>>>(k, xk, AN);
  cvt_f32_to_bf16<<<AN / 1024, 256, 0, stream>>>(v, xv, AN);

  dim3 gg(DDIM / BN, SEQ / BM);  // (8, 32)
  gemm_bf16_wmma<GM_QK><<<gg, 256, 0, stream>>>(xq, wqb, bq, qbf, 0.125f);
  gemm_bf16_wmma<GM_QK><<<gg, 256, 0, stream>>>(xk, wkb, bk, kbf, 1.0f);
  gemm_bf16_wmma<GM_VT><<<gg, 256, 0, stream>>>(xv, wvb, bv, vtb, 1.0f);

  attn_wmma<<<dim3(SEQ / 128, GRP), 256, 0, stream>>>(qbf, kbf, vtb, mask, abf);

  gemm_bf16_wmma<GM_OUT><<<gg, 256, 0, stream>>>(abf, wob, bo, d_out, 1.0f);
}
